// Circulant1x1Conv_4939212390874
// MI455X (gfx1250) — matrix operand order repaired and checked by reading, not compiled, not run on hardware
//
#include <hip/hip_runtime.h>

typedef __attribute__((ext_vector_type(16))) _Float16 v16h;
typedef __attribute__((ext_vector_type(8)))  _Float16 v8h;
typedef __attribute__((ext_vector_type(8)))  float    v8f;
typedef int v4i __attribute__((vector_size(16)));   // matches builtin's param type

#define SIZE_C   512
#define NSTACK   4
#define NBATCH   32
#define HW       1024                 // 32*32 spatial
#define MDIM     (NSTACK * SIZE_C)    // 2048 output channels
#define NFREQ    257

#define M_TILE   128
#define N_TILE   128
#define K_TILE   32
#define LDK      40                   // padded LDS row stride in halves (80 B, 16B-aligned)
#define KITERS   (SIZE_C / K_TILE)    // 16

// ---------------- gfx1250 async global->LDS helpers (builtin if declared, else inline asm) ----
__device__ __forceinline__ void async_b128(const _Float16* g, _Float16* l)
{
#if __has_builtin(__builtin_amdgcn_global_load_async_to_lds_b128)
    __builtin_amdgcn_global_load_async_to_lds_b128(
        (__attribute__((address_space(1))) v4i*)(v4i*)(void*)(_Float16*)g,
        (__attribute__((address_space(3))) v4i*)(v4i*)(void*)l,
        0, 0);
#else
    const unsigned loff =
        (unsigned)(size_t)(__attribute__((address_space(3))) _Float16*)l;
    asm volatile("global_load_async_to_lds_b128 %0, %1, off"
                 :: "v"(loff), "v"(g) : "memory");
#endif
}

__device__ __forceinline__ void wait_async0()
{
#if __has_builtin(__builtin_amdgcn_s_wait_asynccnt)
    __builtin_amdgcn_s_wait_asynccnt(0);
#else
    asm volatile("s_wait_asynccnt 0" ::: "memory");
#endif
}

// ---------------- pass 1: inverse rfft of c_f -> time-domain c (4 x 512, f32) -------------
__global__ void k_irfft(const float* __restrict__ cf, float* __restrict__ c)
{
    const int s = blockIdx.x;       // stack 0..3
    const int n = threadIdx.x;      // 0..511
    const float* p = cf + s * NFREQ * 2;     // (re,im) pairs
    float acc = p[0];                                   // DC (real)
    acc += ((n & 1) ? -1.0f : 1.0f) * p[2 * 256];       // Nyquist (real)
    const float w = 6.283185307179586f * (float)n / 512.0f;
    for (int k = 1; k < 256; ++k) {
        float sn, cs;
        __sincosf(w * (float)k, &sn, &cs);
        acc += 2.0f * (p[2 * k] * cs - p[2 * k + 1] * sn);
    }
    c[s * SIZE_C + n] = acc * (1.0f / 512.0f);
}

// ---------------- pass 2: build A = W^T in f16, A[s*512+n][k] = c_s[(n-k) mod 512] --------
__global__ void k_buildA(const float* __restrict__ c, _Float16* __restrict__ A)
{
    const int m  = blockIdx.x;          // 0..2047
    const int s  = m >> 9;
    const int nn = m & 511;
    const int k  = threadIdx.x << 1;    // 256 threads, 2 elems each
    const float* cs = c + s * SIZE_C;
    _Float16* row = A + (size_t)m * SIZE_C;
    row[k]     = (_Float16)cs[(nn - k)     & 511];
    row[k + 1] = (_Float16)cs[(nn - k - 1) & 511];
}

// ---------------- pass 3: transpose + convert input -> XT[b][n][k] f16 --------------------
__global__ __launch_bounds__(256) void k_xt(const float* __restrict__ x,
                                            _Float16* __restrict__ xt)
{
    __shared__ _Float16 t[32][33];
    const int b  = blockIdx.z;
    const int k0 = blockIdx.y * 32;     // channel tile
    const int n0 = blockIdx.x * 32;     // spatial tile
    const int tx = threadIdx.x & 31;
    const int ty = threadIdx.x >> 5;    // 0..7
    const float* xin = x + ((size_t)b * SIZE_C + k0) * HW + n0;
#pragma unroll
    for (int i = 0; i < 32; i += 8)
        t[ty + i][tx] = (_Float16)xin[(size_t)(ty + i) * HW + tx];   // [k][n]
    __syncthreads();
    _Float16* xo = xt + ((size_t)b * HW + n0) * SIZE_C + k0;
#pragma unroll
    for (int i = 0; i < 32; i += 8)
        xo[(size_t)(ty + i) * SIZE_C + tx] = t[tx][ty + i];          // [n][k]
}

// ---------------- pass 4: per-batch GEMM: out_b = A (2048x512) @ XT_b^T (512x1024) --------
__global__ __launch_bounds__(256)
void k_gemm(const _Float16* __restrict__ Aw,
            const _Float16* __restrict__ XT,
            float* __restrict__ out)
{
    __shared__ _Float16 sA[2][M_TILE * LDK];   // [m][k] halves
    __shared__ _Float16 sB[2][N_TILE * LDK];   // [n][k] halves

    const int b  = blockIdx.z;
    const int m0 = blockIdx.y * M_TILE;
    const int n0 = blockIdx.x * N_TILE;

    const int tid  = threadIdx.x;
    const int lane = tid & 31;
    const int wave = tid >> 5;      // 0..7
    const int wm   = wave >> 2;     // 0..1 : 64-row strip
    const int wn   = wave & 3;      // 0..3 : 32-col strip

    const _Float16* Ag = Aw + (size_t)m0 * SIZE_C;
    const _Float16* Bg = XT + ((size_t)b * HW + n0) * SIZE_C;

    const int ldr = tid >> 1;           // tile row 0..127
    const int seg = (tid & 1) << 4;     // 0 or 16 halves

    v8f zero = {};
    v8f acc[4][2];
#pragma unroll
    for (int i = 0; i < 4; ++i)
#pragma unroll
        for (int j = 0; j < 2; ++j) acc[i][j] = zero;

    // ---- prologue: async-DMA stage 0 into buffer 0 ----
    {
        const _Float16* ga = Ag + (size_t)ldr * SIZE_C + seg;
        const _Float16* gb = Bg + (size_t)ldr * SIZE_C + seg;
        async_b128(ga,     &sA[0][ldr * LDK + seg]);
        async_b128(ga + 8, &sA[0][ldr * LDK + seg + 8]);
        async_b128(gb,     &sB[0][ldr * LDK + seg]);
        async_b128(gb + 8, &sB[0][ldr * LDK + seg + 8]);
    }
    wait_async0();
    __syncthreads();

    const int rsel = (lane < 16) ? 0 : 8;    // A-frag K base (0..7/16..23 vs 8..15/24..31)
    const int csel = (lane < 16) ? 0 : 8;    // B-frag K base (0..7/16..23 vs 8..15/24..31)
    const int rlo  = lane & 15;

    for (int kt = 0; kt < KITERS; ++kt) {
        const int buf = kt & 1;
        if (kt + 1 < KITERS) {
            // buffer buf^1 is free (barrier at end of previous iteration): DMA next stage now
            const int nb = buf ^ 1;
            const int kn = (kt + 1) * K_TILE;
            const _Float16* ga = Ag + (size_t)ldr * SIZE_C + kn + seg;
            const _Float16* gb = Bg + (size_t)ldr * SIZE_C + kn + seg;
            async_b128(ga,     &sA[nb][ldr * LDK + seg]);
            async_b128(ga + 8, &sA[nb][ldr * LDK + seg + 8]);
            async_b128(gb,     &sB[nb][ldr * LDK + seg]);
            async_b128(gb + 8, &sB[nb][ldr * LDK + seg + 8]);
        }

        const _Float16* tA = sA[buf];
        const _Float16* tB = sB[buf];

        v16h aF[4];
#pragma unroll
        for (int fm = 0; fm < 4; ++fm) {
            const int r = wm * 64 + fm * 16 + rlo;
            v8h lo = *(const v8h*)(tA + r * LDK + rsel);
            v8h hi = *(const v8h*)(tA + r * LDK + rsel + 16);
            aF[fm] = __builtin_shufflevector(lo, hi, 0,1,2,3,4,5,6,7,8,9,10,11,12,13,14,15);
        }
        v16h bF[2];
#pragma unroll
        for (int fn = 0; fn < 2; ++fn) {
            const int cI = wn * 32 + fn * 16 + rlo;
            v8h lo = *(const v8h*)(tB + cI * LDK + csel);
            v8h hi = *(const v8h*)(tB + cI * LDK + csel + 16);
            bF[fn] = __builtin_shufflevector(lo, hi, 0,1,2,3,4,5,6,7,8,9,10,11,12,13,14,15);
        }

#pragma unroll
        for (int fm = 0; fm < 4; ++fm)
#pragma unroll
            for (int fn = 0; fn < 2; ++fn)
                acc[fm][fn] = __builtin_amdgcn_wmma_f32_16x16x32_f16(
                    false, aF[fm], false, bF[fn],
                    (short)0, acc[fm][fn], false, false);

        if (kt + 1 < KITERS) {
            wait_async0();      // this wave's DMA into buf^1 has landed
            __syncthreads();    // everyone's DMA visible; buf free for next stage
        }
    }

    // ---- epilogue: C/D layout — VGPR j: lanes 0-15 -> M=j, lanes 16-31 -> M=j+8 ----
    float* Cb = out + ((size_t)b * MDIM + m0) * HW + n0;
    const int rup = (lane < 16) ? 0 : 8;
#pragma unroll
    for (int fm = 0; fm < 4; ++fm)
#pragma unroll
        for (int fn = 0; fn < 2; ++fn) {
            const int rbase = wm * 64 + fm * 16 + rup;
            const int col   = wn * 32 + fn * 16 + rlo;
            v8f a = acc[fm][fn];
#pragma unroll
            for (int j = 0; j < 8; ++j)
                Cb[(size_t)(rbase + j) * HW + col] = a[j];
        }
}

extern "C" void kernel_launch(void* const* d_in, const int* in_sizes, int n_in,
                              void* d_out, int out_size, void* d_ws, size_t ws_size,
                              hipStream_t stream)
{
    const float* x  = (const float*)d_in[0];   // (32, 512, 32, 32) f32
    const float* cf = (const float*)d_in[1];   // (4, 257, 2) f32
    float* out = (float*)d_out;                // (32, 2048, 32, 32) f32

    char* ws = (char*)d_ws;
    float*    c_time = (float*)ws;                                          // 8 KB
    _Float16* Aw     = (_Float16*)(ws + 8192);                              // 2 MiB
    _Float16* XT     = (_Float16*)(ws + 8192 + (size_t)MDIM * SIZE_C * 2);  // 32 MiB

    k_irfft <<<dim3(NSTACK),                       dim3(SIZE_C), 0, stream>>>(cf, c_time);
    k_buildA<<<dim3(MDIM),                         dim3(256),    0, stream>>>(c_time, Aw);
    k_xt    <<<dim3(HW / 32, SIZE_C / 32, NBATCH), dim3(256),    0, stream>>>(x, XT);
    k_gemm  <<<dim3(HW / N_TILE, MDIM / M_TILE, NBATCH), dim3(256), 0, stream>>>(Aw, XT, out);
}
